// ProtoPNetConv_81810537054613
// MI455X (gfx1250) — matrix operand order repaired and checked by reading, not compiled, not run on hardware
//
#include <hip/hip_runtime.h>
#include <math.h>

// Problem constants (from reference)
#define ENC    1024
#define PROTO_K   4
#define NPROT   100
#define BATCH    32
#define SEQ     512
#define SEQ_P   528   // padded rows (zeros) so dilated taps never go OOB
#define CDIM   4096   // ENC * PROTO_K, contraction length
#define MPAD     48   // padded prototype rows per dilation group (3 x 16)

// ---- CDNA5 WMMA / TDM types ----
typedef __attribute__((ext_vector_type(16))) __bf16 v16bf;
typedef __attribute__((ext_vector_type(8)))  float  v8f;
typedef __attribute__((ext_vector_type(4)))  unsigned int u32x4;
typedef __attribute__((ext_vector_type(8)))  int i32x8;
typedef __attribute__((ext_vector_type(4)))  int i32x4;

union AB16 { v16bf v; unsigned int u[8]; };

__device__ __forceinline__ unsigned short f2bf(float f) {
  // round-to-nearest-even f32 -> bf16
  unsigned int u = __float_as_uint(f);
  u += 0x7FFFu + ((u >> 16) & 1u);
  return (unsigned short)(u >> 16);
}

// Issue one 2-D bf16 tile load via the Tensor Data Mover.
// D# per CDNA5 ISA: group0 = {flags, lds_addr,
// global_addr[56:0], type=2}; group1 = {data_size=1(2B), tensor_dim0=32,
// tensor_dim1=rows, tile_dim0=32, tile_dim1=rows, dim0_stride=row_stride}.
// groups 2/3 zero (2-D tile: tile_dim2=0 -> unused). Tracked by TENSORcnt.
// This toolchain's builtin takes 6 args (extra i32x8 before cpol).
__device__ __forceinline__ void tdm_load_2d(unsigned lds_off, const void* gaddr,
                                            unsigned rows, unsigned row_stride) {
  unsigned long long ga = (unsigned long long)(size_t)gaddr;
  u32x4 g0;
  g0[0] = 1u;                                   // count=1 (valid user D#)
  g0[1] = lds_off;                              // lds_addr (bytes)
  g0[2] = (unsigned)(ga & 0xFFFFFFFFu);         // global_addr[31:0]
  g0[3] = (unsigned)(ga >> 32) | (2u << 30);    // global_addr[56:32] | type=2
  i32x8 g1;
  g1[0] = 0x00010000;                           // data_size=1 (2B); no flags
  g1[1] = (int)(32u << 16);                     // tensor_dim0 = 32 (bits 63:48)
  g1[2] = (int)(rows << 16);                    // tensor_dim1[15:0] (bits 95:80)
  g1[3] = (int)(32u << 16);                     // tile_dim0 = 32 (bits 127:112)
  g1[4] = (int)rows;                            // tile_dim1 (bits 143:128)
  g1[5] = (int)row_stride;                      // tensor_dim0_stride[31:0]
  g1[6] = 0;
  g1[7] = 0;
  i32x4 z4 = {0, 0, 0, 0};
  i32x8 z8 = {0, 0, 0, 0, 0, 0, 0, 0};
  __builtin_amdgcn_tensor_load_to_lds(g0, g1, z4, z4, z8, 0);
}

__device__ __forceinline__ void group_params(int g, int& dil, int& n, int& j) {
  dil = 1 << g;                       // DILATED = {1,2,4}
  n   = (g == 0) ? 34 : 33;           // NUM_FILTERS = {34,33,33}
  j   = (g == 0) ? 0 : (g == 1 ? 34 : 67);
}

// ---------------------------------------------------------------------------
// out[0:3200) holds prototype_distances as float bits; seed with +INF so
// bitwise atomicMin over non-negative floats computes the true min.
__global__ void init_out_kernel(unsigned int* __restrict__ ob) {
  int i = blockIdx.x * blockDim.x + threadIdx.x;
  if (i < BATCH * NPROT) ob[i] = 0x7F800000u;
}

// A-pack: Apk[g][m][c] = bf16(protolayer[j+m][c]) (c = k_tap*1024 + e is the
// native protolayer order), zero padding rows m >= n.
__global__ void pack_protos_kernel(const float* __restrict__ proto,
                                   unsigned short* __restrict__ Apk) {
  int g = blockIdx.x / MPAD;
  int m = blockIdx.x % MPAD;
  int dil, n, j; group_params(g, dil, n, j);
  unsigned short* dst = Apk + ((size_t)g * MPAD + m) * CDIM;
  const float*    src = proto + (size_t)(j + m) * CDIM;
  for (int c = threadIdx.x; c < CDIM; c += blockDim.x) {
    float v = (m < n) ? src[c] : 0.0f;
    dst[c] = f2bf(v);
  }
}

// Embedding pack: Epk[b][l][e] = bf16(emb[b][l][e]), rows l in [512,528) = 0.
// One-time 64MB->32MB conversion so the hot loop streams bf16 via TDM.
__global__ void pack_emb_kernel(const float* __restrict__ emb,
                                unsigned short* __restrict__ Epk) {
  int bl = blockIdx.x;                    // 0 .. BATCH*SEQ_P-1
  int b = bl / SEQ_P, l = bl % SEQ_P;
  unsigned short* dst = Epk + (size_t)bl * ENC;
  int t = threadIdx.x;                    // 256 threads x 4 elems = ENC
  unsigned short tmp[4];
  if (l < SEQ) {
    float4 f = ((const float4*)(emb + ((size_t)b * SEQ + l) * ENC))[t];
    tmp[0] = f2bf(f.x); tmp[1] = f2bf(f.y); tmp[2] = f2bf(f.z); tmp[3] = f2bf(f.w);
  } else {
    tmp[0] = tmp[1] = tmp[2] = tmp[3] = 0;
  }
  *(uint2*)&dst[t * 4] = *(const uint2*)tmp;
}

// p2[p] = sum(protolayer[p]^2)
__global__ void p2_kernel(const float* __restrict__ proto, float* __restrict__ p2) {
  __shared__ float red[256];
  int p = blockIdx.x;
  float s = 0.0f;
  for (int c = threadIdx.x; c < CDIM; c += 256) {
    float v = proto[(size_t)p * CDIM + c];
    s += v * v;
  }
  red[threadIdx.x] = s; __syncthreads();
  for (int st = 128; st > 0; st >>= 1) {
    if (threadIdx.x < st) red[threadIdx.x] += red[threadIdx.x + st];
    __syncthreads();
  }
  if (threadIdx.x == 0) p2[p] = red[0];
}

// S[b*SEQ+l] = sum_e emb[b][l][e]^2  -> the "ones filter" conv becomes 4 adds.
__global__ void rowsq_kernel(const float* __restrict__ emb, float* __restrict__ S) {
  __shared__ float red[256];
  size_t bl = blockIdx.x;
  float4 f = ((const float4*)(emb + bl * ENC))[threadIdx.x]; // 256*4 == ENC
  float s = f.x * f.x + f.y * f.y + f.z * f.z + f.w * f.w;
  red[threadIdx.x] = s; __syncthreads();
  for (int st = 128; st > 0; st >>= 1) {
    if (threadIdx.x < st) red[threadIdx.x] += red[threadIdx.x + st];
    __syncthreads();
  }
  if (threadIdx.x == 0) S[bl] = red[0];
}

// ---------------------------------------------------------------------------
// Main conv-as-GEMM kernel. Grid (4 o-chunks x 3 groups x 32 batch), 256 thr
// = 8 waves. Each wave: one 16-col o-tile, three 16-row proto tiles; K=4096
// in 128 steps of v_wmma_f32_16x16x32_bf16 with double-buffered TDM staging
// (wave 0 issues two tensor_load_to_lds per step, everyone else just
// barriers; operands pre-packed to bf16 in workspace).
__global__ __launch_bounds__(256)
void proto_wmma_kernel(const unsigned short* __restrict__ Epk,
                       const unsigned short* __restrict__ Apk,
                       const float* __restrict__ p2,
                       const float* __restrict__ S,
                       unsigned int* __restrict__ out_bits) {
  const int oc = blockIdx.x;           // o-chunk: 128 positions
  const int g  = blockIdx.y;           // dilation group
  const int b  = blockIdx.z;           // batch
  int dil, n, j; group_params(g, dil, n, j);
  const int Lo = SEQ - 3 * dil;
  const int obase = oc * 128;

  __shared__ __align__(16) unsigned short as_s[2][MPAD][32]; // A chunks (bf16)
  __shared__ __align__(16) unsigned short xs_s[2][128][32];  // B chunks (bf16)

  const int tid  = threadIdx.x;
  const int lane = tid & 31;
  const int wv   = tid >> 5;           // wave id 0..7 -> o sub-tile
  const int h    = lane >> 4;          // lane half (wave32 WMMA layout)
  const int col  = lane & 15;

  v8f acc0 = {}; v8f acc1 = {}; v8f acc2 = {};

  const unsigned short* apg  = Apk + (size_t)g * MPAD * CDIM;
  const unsigned short* epkb = Epk + (size_t)b * SEQ_P * ENC;

  const unsigned lds_as0 = (unsigned)(size_t)&as_s[0][0][0];
  const unsigned lds_as1 = (unsigned)(size_t)&as_s[1][0][0];
  const unsigned lds_xs0 = (unsigned)(size_t)&xs_s[0][0][0];
  const unsigned lds_xs1 = (unsigned)(size_t)&xs_s[1][0][0];

  // Stage one 32-wide contraction chunk into LDS buffer `buf` via TDM.
  auto stage = [&](int kc, int buf) {
    const int ktap = kc >> 5;          // conv tap 0..3
    const int e0   = (kc & 31) << 5;   // e offset within tap
    tdm_load_2d(buf ? lds_as1 : lds_as0,
                apg + ktap * 1024 + e0, MPAD, CDIM);
    tdm_load_2d(buf ? lds_xs1 : lds_xs0,
                epkb + (size_t)(obase + dil * ktap) * ENC + e0, 128, ENC);
  };

  if (wv == 0) {
    stage(0, 0);
    __builtin_amdgcn_s_wait_tensorcnt(0);
  }
  __syncthreads();

  for (int kc = 0; kc < 128; ++kc) {   // 128 * 32 = 4096 contraction
    const int cur = kc & 1;
    if (wv == 0 && kc + 1 < 128) stage(kc + 1, cur ^ 1);  // prefetch next

    // Gather WMMA fragments per the documented gfx1250 16-bit layouts.
    AB16 a0, a1, a2, bf;
#pragma unroll
    for (int v = 0; v < 8; ++v) {
      // A 16x32: lanes 0-15 K=0..7 (v0-3), 16..23 (v4-7); lanes 16-31 +8
      int Ka = ((v & 3) * 2) + ((v >> 2) * 16) + h * 8;
      a0.u[v] = *(const unsigned int*)&as_s[cur][col     ][Ka];
      a1.u[v] = *(const unsigned int*)&as_s[cur][16 + col][Ka];
      a2.u[v] = *(const unsigned int*)&as_s[cur][32 + col][Ka];
      // B 32x16: same K order as A: lanes 0-15 K=0..7 (v0-3), 16..23 (v4-7); lanes 16-31 +8
      int Kb = ((v & 3) * 2) + ((v >> 2) * 16) + h * 8;
      bf.u[v] = *(const unsigned int*)&xs_s[cur][wv * 16 + col][Kb];
    }
    acc0 = __builtin_amdgcn_wmma_f32_16x16x32_bf16(false, a0.v, false, bf.v,
                                                   (short)0, acc0, false, false);
    acc1 = __builtin_amdgcn_wmma_f32_16x16x32_bf16(false, a1.v, false, bf.v,
                                                   (short)0, acc1, false, false);
    acc2 = __builtin_amdgcn_wmma_f32_16x16x32_bf16(false, a2.v, false, bf.v,
                                                   (short)0, acc2, false, false);

    if (wv == 0) __builtin_amdgcn_s_wait_tensorcnt(0);  // next buffer landed
    __syncthreads();   // publish TDM-written LDS + finish reads of cur
  }

  // Epilogue: dist = S2(o) - 2*xp + p2, sqrt|.|, min over o-tile, atomicMin.
  const int o = obase + wv * 16 + col;
  float s2 = 0.0f;
#pragma unroll
  for (int kt = 0; kt < 4; ++kt) {
    int l = o + dil * kt;
    s2 += (l < SEQ) ? S[b * SEQ + l] : 0.0f;
  }
  const float INFV = __uint_as_float(0x7F800000u);
#pragma unroll
  for (int mt = 0; mt < 3; ++mt) {
    v8f acc = (mt == 0) ? acc0 : ((mt == 1) ? acc1 : acc2);
#pragma unroll
    for (int r = 0; r < 8; ++r) {
      // C 16x16 f32: lanes 0-15 -> M=r, lanes 16-31 -> M=r+8, N=col
      int mloc = mt * 16 + r + h * 8;
      float pq = (mloc < n) ? p2[j + mloc] : 0.0f;
      float d  = s2 - 2.0f * acc[r] + pq;
      float val = sqrtf(fabsf(d));
      if (o >= Lo) val = INFV;
      // min over the 16 columns sharing this M
      #pragma unroll
      for (int off = 8; off > 0; off >>= 1)
        val = fminf(val, __shfl_xor(val, off, 16));
      if (col == 0 && mloc < n)
        atomicMin(&out_bits[b * NPROT + j + mloc], __float_as_uint(val));
    }
  }
}

// class_out = pd @ fc_w^T  ([32,100] x [100,2])
__global__ void fc_kernel(const float* __restrict__ pd,
                          const float* __restrict__ fcw,
                          float* __restrict__ out_cls) {
  int t = threadIdx.x;
  if (t < BATCH * 2) {
    int b = t >> 1, c = t & 1;
    float s = 0.0f;
    for (int p = 0; p < NPROT; ++p)
      s += pd[b * NPROT + p] * fcw[c * NPROT + p];
    out_cls[b * 2 + c] = s;
  }
}

// ---------------------------------------------------------------------------
extern "C" void kernel_launch(void* const* d_in, const int* in_sizes, int n_in,
                              void* d_out, int out_size, void* d_ws, size_t ws_size,
                              hipStream_t stream) {
  (void)in_sizes; (void)n_in; (void)out_size; (void)ws_size;
  const float* emb   = (const float*)d_in[0];  // [32,512,1024] f32
  const float* proto = (const float*)d_in[1];  // [100,4,1024] f32
  const float* fcw   = (const float*)d_in[2];  // [2,100] f32
  float* out = (float*)d_out;                  // 3200 pd + 64 class

  // workspace layout (~35.9 MB)
  unsigned short* Apk = (unsigned short*)d_ws;                    // 1,179,648 B
  float* p2 = (float*)((char*)d_ws + 1179648);                    //   400 B
  float* S  = (float*)((char*)d_ws + 1179648 + 512);              // 65,536 B
  unsigned short* Epk = (unsigned short*)((char*)d_ws + 1245696); // 34,603,008 B

  init_out_kernel<<<dim3(13), dim3(256), 0, stream>>>((unsigned int*)d_out);
  pack_protos_kernel<<<dim3(3 * MPAD), dim3(256), 0, stream>>>(proto, Apk);
  pack_emb_kernel<<<dim3(BATCH * SEQ_P), dim3(256), 0, stream>>>(emb, Epk);
  p2_kernel<<<dim3(NPROT), dim3(256), 0, stream>>>(proto, p2);
  rowsq_kernel<<<dim3(BATCH * SEQ), dim3(256), 0, stream>>>(emb, S);
  proto_wmma_kernel<<<dim3(4, 3, BATCH), dim3(256), 0, stream>>>(
      Epk, Apk, p2, S, (unsigned int*)d_out);
  fc_kernel<<<dim3(1), dim3(64), 0, stream>>>(out, fcw, out + BATCH * NPROT);
}
